// FunASREncoder_61546881352166
// MI455X (gfx1250) — compile-verified, matrix operand order checked by reading, not run on hardware
//
#include <hip/hip_runtime.h>
#include <cstdint>
#include <cstddef>

// ---------------------------------------------------------------------------
// Types for CDNA5 WMMA (wave32). bf16 A/B fragments, fp32 accumulator.
// ---------------------------------------------------------------------------
typedef __attribute__((ext_vector_type(16))) __bf16 v16bf;
typedef __attribute__((ext_vector_type(8)))  __bf16 v8bf;
typedef __attribute__((ext_vector_type(8)))  float  v8f;
typedef __attribute__((ext_vector_type(4)))  unsigned int u32x4;
typedef __attribute__((ext_vector_type(8)))  int i32x8;
typedef __attribute__((ext_vector_type(4)))  int i32x4;

#define N_B    8
#define N_T    1000
#define N_DIN  560
#define N_D    512
#define N_H    4
#define N_DK   128
#define N_F    2048
#define N_KC   11
#define N_TD3  1536      // 3*D
#define N_M    8000      // B*T
#define N_SLD  1024      // padded key length for PV WMMA (mult of 32)
#define N_QT   63        // ceil(T/16)
#define N_KPAD0 576      // DIN=560 padded to mult of 32

// TDM path available? (this toolchain: 6-arg builtin, clang-23/therock form)
#if defined(__HIP_DEVICE_COMPILE__) && __has_builtin(__builtin_amdgcn_tensor_load_to_lds)
#define GEMM_TDM 1
#else
#define GEMM_TDM 0
#endif

__device__ __forceinline__ __bf16 f2bf(float f) {
  union { float f; uint32_t u; } in; in.f = f;
  uint32_t u = in.u + 0x7FFFu + ((in.u >> 16) & 1u);  // RNE
  union { uint16_t s; __bf16 b; } out; out.s = (uint16_t)(u >> 16);
  return out.b;
}
__device__ __forceinline__ __bf16 bf0() {
  union { uint16_t s; __bf16 b; } out; out.s = 0; return out.b;
}
__device__ __forceinline__ v8f wmma_bf16(v16bf a, v16bf b, v8f c) {
  // (neg_a, A, neg_b, B, c_mod, C, reuse_a, reuse_b)
  return __builtin_amdgcn_wmma_f32_16x16x32_bf16(false, a, false, b, (short)0, c,
                                                 false, false);
}
#define SHUF16(lo, hi) \
  __builtin_shufflevector(lo, hi, 0,1,2,3,4,5,6,7,8,9,10,11,12,13,14,15)

// ---------------------------------------------------------------------------
// posenc + input scaling: x = xs*sqrt(D) + PE(t,d)
// ---------------------------------------------------------------------------
__global__ void k_posenc(const float* __restrict__ xs, float* __restrict__ o) {
  size_t i = (size_t)blockIdx.x * blockDim.x + threadIdx.x;
  if (i >= (size_t)N_B * N_T * N_DIN) return;
  int d = (int)(i % N_DIN);
  int t = (int)((i / N_DIN) % N_T);
  const int half = N_DIN / 2;
  int dd = (d < half) ? d : d - half;
  float inv = __expf(-(float)dd * (logf(10000.0f) / (float)(half - 1)));
  float st  = (float)(t + 1) * inv;
  float pe  = (d < half) ? __sinf(st) : __cosf(st);
  o[i] = xs[i] * 22.62741699796952f + pe;  // sqrt(512)
}

// ---------------------------------------------------------------------------
// LayerNorm: fp32 in -> optional bf16 out (K padded with zeros), optional fp32 out
// One block (256 thr) per row; row cached in registers -> in-place safe.
// ---------------------------------------------------------------------------
__global__ __launch_bounds__(256) void k_ln(
    const float* __restrict__ x, const float* __restrict__ sc,
    const float* __restrict__ bi, __bf16* __restrict__ out_bf,
    float* __restrict__ out_f, int Dd, int Kpad) {
  __shared__ float rs[256], rq[256];
  const int row = blockIdx.x;
  const float* xr = x + (size_t)row * Dd;
  float vals[3];
  int cnt = 0;
  float s = 0.f, q = 0.f;
  for (int d = threadIdx.x; d < Dd; d += 256) {
    float v = xr[d];
    vals[cnt++] = v;
    s += v; q += v * v;
  }
  rs[threadIdx.x] = s; rq[threadIdx.x] = q;
  __syncthreads();
  for (int st = 128; st > 0; st >>= 1) {
    if (threadIdx.x < st) {
      rs[threadIdx.x] += rs[threadIdx.x + st];
      rq[threadIdx.x] += rq[threadIdx.x + st];
    }
    __syncthreads();
  }
  float mean = rs[0] / (float)Dd;
  float var  = rq[0] / (float)Dd - mean * mean;
  float rstd = rsqrtf(var + 1e-12f);
  cnt = 0;
  for (int d = threadIdx.x; d < Dd; d += 256) {
    float y = (vals[cnt++] - mean) * rstd * sc[d] + bi[d];
    if (out_bf) out_bf[(size_t)row * Kpad + d] = f2bf(y);
    if (out_f)  out_f[(size_t)row * Dd + d] = y;
  }
  if (out_bf)
    for (int d = Dd + threadIdx.x; d < Kpad; d += 256)
      out_bf[(size_t)row * Kpad + d] = bf0();
}

// ---------------------------------------------------------------------------
// Elementwise fp32 -> bf16
// ---------------------------------------------------------------------------
__global__ void k_f2b(const float* __restrict__ in, __bf16* __restrict__ o,
                      size_t count) {
  size_t i = (size_t)blockIdx.x * blockDim.x + threadIdx.x;
  if (i < count) o[i] = f2bf(in[i]);
}

// Weight convert: fp32 [N,Kin] row-major -> bf16 [N,Kpad] zero padded
__global__ void k_convw(const float* __restrict__ w, __bf16* __restrict__ o,
                        int N, int Kin, int Kpad) {
  size_t i = (size_t)blockIdx.x * blockDim.x + threadIdx.x;
  if (i >= (size_t)N * Kpad) return;
  int k = (int)(i % Kpad);
  int n = (int)(i / Kpad);
  o[i] = (k < Kin) ? f2bf(w[(size_t)n * Kin + k]) : bf0();
}

// ---------------------------------------------------------------------------
// Tiled bf16 WMMA GEMM: out[M,N] = A[M,K] * W[N,K]^T + bias (+add0)(+add1)(relu)
// Block 128 thr = 4 waves, tile 64x32, K-step 32, LDS rows padded to 40 elems
// (80 B) to spread banks.
//  - A tile (64x32): per-lane 16B chunks via GLOBAL_LOAD_ASYNC_TO_LDS_B128
//    (ASYNCcnt), one chunk pair per thread, s_wait_asynccnt before use.
//  - B tile (32Nx32K): Tensor Data Mover 2-D descriptor issued by wave 0;
//    TDM pad feature (16-DWORD interval, 4-DWORD pad) reproduces the
//    40-element LDS row stride. s_wait_tensorcnt 0 + barrier publishes it.
// Requires M%64==0, N%32==0, K%32==0 (guaranteed by padding).
// ---------------------------------------------------------------------------
__global__ __launch_bounds__(128) void k_gemm(
    const __bf16* __restrict__ A, const __bf16* __restrict__ W,
    const float* __restrict__ bias, const float* __restrict__ add0,
    const float* __restrict__ add1, float* __restrict__ out,
    int N, int K, int relu) {
  __shared__ __bf16 As[64 * 40];
  __shared__ __bf16 Bs[32 * 40];
  const int tid  = threadIdx.x;
  const int wave = tid >> 5;
  const int lane = tid & 31;
  const int r16  = lane & 15;
  const int koff = (lane >> 4) << 3;
  const int m0 = blockIdx.y * 64;
  const int n0 = blockIdx.x * 32;
  // A-tile chunk assignment: chunk c -> row c>>2, col ((c&3)<<3); 2 chunks/thr
  const int ra0 = tid >> 2,        cc0 = (tid & 3) << 3;
  const int ra1 = (tid + 128) >> 2, cc1 = ((tid + 128) & 3) << 3;
  const unsigned ldsA0 = (unsigned)(size_t)&As[ra0 * 40 + cc0];
  const unsigned ldsA1 = (unsigned)(size_t)&As[ra1 * 40 + cc1];
  v8f c0 = {}; v8f c1 = {};
  for (int k0 = 0; k0 < K; k0 += 32) {
    __syncthreads();
    {
      unsigned long long gA0 =
          (unsigned long long)(size_t)&A[(size_t)(m0 + ra0) * K + k0 + cc0];
      unsigned long long gA1 =
          (unsigned long long)(size_t)&A[(size_t)(m0 + ra1) * K + k0 + cc1];
      asm volatile("global_load_async_to_lds_b128 %0, %1, off"
                   :: "v"(ldsA0), "v"(gA0) : "memory");
      asm volatile("global_load_async_to_lds_b128 %0, %1, off"
                   :: "v"(ldsA1), "v"(gA1) : "memory");
    }
#if GEMM_TDM
    if (wave == 0) {
      unsigned ldsB = (unsigned)(size_t)&Bs[0];
      unsigned long long gaB =
          (unsigned long long)(size_t)&W[(size_t)n0 * K + k0];
      u32x4 g0;
      g0.x = 1u;                                   // count=1 (valid descriptor)
      g0.y = ldsB;                                 // lds_addr
      g0.z = (unsigned)gaB;                        // global_addr[31:0]
      g0.w = (unsigned)((gaB >> 32) & 0x01FFFFFFull) | (2u << 30);  // ga hi | type=2
      i32x8 g1;
      // data_size=1(2B) | pad_enable | pad_interval=3(16 DW) | pad_amount=3(4 DW)
      g1[0] = (int)((1u << 16) | (1u << 20) | (3u << 22) | (3u << 25));
      g1[1] = (int)(((unsigned)K & 0xFFFFu) << 16);                    // tensor_dim0 lo16
      g1[2] = (int)((((unsigned)K >> 16) & 0xFFFFu) |
                    (((unsigned)N & 0xFFFFu) << 16));                  // dim0 hi | dim1 lo
      g1[3] = (int)((((unsigned)N >> 16) & 0xFFFFu) | (32u << 16));    // dim1 hi | tile_dim0=32
      g1[4] = 32;                                                      // tile_dim1=32, tile_dim2=0
      g1[5] = (int)(unsigned)K;                                        // tensor_dim0_stride lo32
      g1[6] = 0;                                                       // stride hi | dim1_stride lo
      g1[7] = 0;
      i32x4 g2 = {0, 0, 0, 0};
      i32x4 g3 = {0, 0, 0, 0};
      i32x8 g4 = {0, 0, 0, 0, 0, 0, 0, 0};
      __builtin_amdgcn_tensor_load_to_lds(g0, g1, g2, g3, g4, 0);
      __builtin_amdgcn_s_wait_tensorcnt(0);
    }
#else
    {
      int row = tid >> 2, cc = (tid & 3) << 3;
      *(uint4*)&Bs[row * 40 + cc] =
          *(const uint4*)&W[(size_t)(n0 + row) * K + k0 + cc];
    }
#endif
    asm volatile("s_wait_asynccnt 0" ::: "memory");
    __syncthreads();
    v8bf alo = *(const v8bf*)&As[(wave * 16 + r16) * 40 + koff];
    v8bf ahi = *(const v8bf*)&As[(wave * 16 + r16) * 40 + koff + 16];
    v16bf a  = SHUF16(alo, ahi);
    v8bf b0lo = *(const v8bf*)&Bs[r16 * 40 + koff];
    v8bf b0hi = *(const v8bf*)&Bs[r16 * 40 + koff + 16];
    v16bf b0  = SHUF16(b0lo, b0hi);
    v8bf b1lo = *(const v8bf*)&Bs[(16 + r16) * 40 + koff];
    v8bf b1hi = *(const v8bf*)&Bs[(16 + r16) * 40 + koff + 16];
    v16bf b1  = SHUF16(b1lo, b1hi);
    c0 = wmma_bf16(a, b0, c0);
    c1 = wmma_bf16(a, b1, c1);
  }
  const int gm0 = m0 + wave * 16 + (lane >> 4) * 8;
#pragma unroll
  for (int v = 0; v < 8; ++v) {
    int gm = gm0 + v;
    int gn = n0 + r16;
    size_t i0 = (size_t)gm * N + gn;
    float v0 = c0[v] + bias[gn];
    if (add0) v0 += add0[i0];
    if (add1) v0 += add1[i0];
    if (relu) v0 = fmaxf(v0, 0.f);
    out[i0] = v0;
    size_t i1 = i0 + 16;
    float v1 = c1[v] + bias[gn + 16];
    if (add0) v1 += add0[i1];
    if (add1) v1 += add1[i1];
    if (relu) v1 = fmaxf(v1, 0.f);
    out[i1] = v1;
  }
}

// ---------------------------------------------------------------------------
// FSMN depthwise conv branch: fsmn = (conv(v*mask) + v*mask) * mask
// v = qkv[:, 2D:3D]
// ---------------------------------------------------------------------------
__global__ void k_fsmn(const float* __restrict__ qkv, const float* __restrict__ w,
                       const int* __restrict__ ilens, float* __restrict__ out) {
  size_t i = (size_t)blockIdx.x * blockDim.x + threadIdx.x;
  if (i >= (size_t)N_M * N_D) return;
  int d   = (int)(i % N_D);
  int row = (int)(i / N_D);
  int t = row % N_T, b = row / N_T;
  int L = ilens[b];
  float r = 0.f;
  if (t < L) {
#pragma unroll
    for (int j = 0; j < N_KC; ++j) {
      int tau = t + j - 5;  // lp = (K-1)/2 = 5, cross-correlation
      if (tau >= 0 && tau < L)
        r += w[d * N_KC + j] *
             qkv[((size_t)(b * N_T + tau)) * N_TD3 + 2 * N_D + d];
    }
    r += qkv[(size_t)row * N_TD3 + 2 * N_D + d];
  }
  out[i] = r;
}

// ---------------------------------------------------------------------------
// Attention scores: one wave per 16x16 (q,k) tile. S = Q K^T / sqrt(DK)
// qkv bf16 layout [B*T, 1536]; scores fp32 [B*H, T, SLD]
// Fragments are contiguous 2x16B chunks -> vector loads.
// ---------------------------------------------------------------------------
__global__ __launch_bounds__(32) void k_scores(
    const __bf16* __restrict__ qkv, float* __restrict__ sc) {
  const int qt = blockIdx.x, kt = blockIdx.y, bh = blockIdx.z;
  const int b = bh >> 2, h = bh & 3;
  const int lane = threadIdx.x;
  const int r16  = lane & 15;
  const int koff = (lane >> 4) << 3;
  const int qrow = qt * 16 + r16;
  const int kcol = kt * 16 + r16;
  const bool qv = qrow < N_T, kv = kcol < N_T;
  const size_t qbase = ((size_t)(b * N_T + qrow)) * N_TD3 + h * N_DK;
  const size_t kbase = ((size_t)(b * N_T + kcol)) * N_TD3 + N_D + h * N_DK;
  v8f c = {};
  for (int k0 = 0; k0 < N_DK; k0 += 32) {
    v16bf a = {}, bb = {};
    if (qv) {
      v8bf lo = *(const v8bf*)&qkv[qbase + k0 + koff];
      v8bf hi = *(const v8bf*)&qkv[qbase + k0 + koff + 16];
      a = SHUF16(lo, hi);
    }
    if (kv) {
      v8bf lo = *(const v8bf*)&qkv[kbase + k0 + koff];
      v8bf hi = *(const v8bf*)&qkv[kbase + k0 + koff + 16];
      bb = SHUF16(lo, hi);
    }
    c = wmma_bf16(a, bb, c);
  }
  const float scale = 0.08838834764831845f;  // 1/sqrt(128)
  const int m0 = (lane >> 4) * 8;
  float* srow = sc + ((size_t)bh * N_T) * N_SLD;
#pragma unroll
  for (int v = 0; v < 8; ++v) {
    int q = qt * 16 + m0 + v;
    if (q < N_T && kcol < N_T)
      srow[(size_t)q * N_SLD + kcol] = c[v] * scale;
  }
}

// ---------------------------------------------------------------------------
// Masked softmax per (b,h,q) row; writes bf16 probabilities, zero padding.
// ---------------------------------------------------------------------------
__global__ __launch_bounds__(256) void k_softmax(
    const float* __restrict__ sc, const int* __restrict__ ilens,
    __bf16* __restrict__ pbf) {
  const int q = blockIdx.x, bh = blockIdx.y, b = bh >> 2;
  const int L = ilens[b];
  const float* row = sc + ((size_t)bh * N_T + q) * N_SLD;
  __bf16* prow = pbf + ((size_t)bh * N_T + q) * N_SLD;
  __shared__ float red[256];
  float mx = -3.0e38f;
  for (int k = threadIdx.x; k < L; k += 256) mx = fmaxf(mx, row[k]);
  red[threadIdx.x] = mx; __syncthreads();
  for (int st = 128; st > 0; st >>= 1) {
    if (threadIdx.x < st)
      red[threadIdx.x] = fmaxf(red[threadIdx.x], red[threadIdx.x + st]);
    __syncthreads();
  }
  mx = red[0]; __syncthreads();
  float s = 0.f;
  for (int k = threadIdx.x; k < L; k += 256) s += __expf(row[k] - mx);
  red[threadIdx.x] = s; __syncthreads();
  for (int st = 128; st > 0; st >>= 1) {
    if (threadIdx.x < st) red[threadIdx.x] += red[threadIdx.x + st];
    __syncthreads();
  }
  float inv = 1.0f / red[0];
  for (int k = threadIdx.x; k < N_SLD; k += 256) {
    float p = (k < L) ? __expf(row[k] - mx) * inv : 0.f;
    prow[k] = f2bf(p);
  }
}

// ---------------------------------------------------------------------------
// O = P * V : one wave per 16(q) x 16(d) tile; K-loop over padded 1024 keys.
// P fragment contiguous -> vector loads; V fragment column-strided -> scalar.
// ---------------------------------------------------------------------------
__global__ __launch_bounds__(32) void k_av(
    const __bf16* __restrict__ pbf, const __bf16* __restrict__ qkv,
    float* __restrict__ ao) {
  const int qt = blockIdx.x, dt = blockIdx.y, bh = blockIdx.z;
  const int b = bh >> 2, h = bh & 3;
  const int lane = threadIdx.x;
  const int r16  = lane & 15;
  const int koff = (lane >> 4) << 3;
  const int qrow = qt * 16 + r16;
  const bool qv = qrow < N_T;
  const __bf16* prow = pbf + ((size_t)bh * N_T + qrow) * N_SLD;
  const size_t vcol = 2 * N_D + (size_t)h * N_DK + dt * 16 + r16;
  const __bf16 z = bf0();
  v8f c = {};
  for (int k0 = 0; k0 < N_SLD; k0 += 32) {
    v16bf a = {}, bb;
    if (qv) {
      v8bf lo = *(const v8bf*)&prow[k0 + koff];
      v8bf hi = *(const v8bf*)&prow[k0 + koff + 16];
      a = SHUF16(lo, hi);
    }
#pragma unroll
    for (int e = 0; e < 16; ++e) {
      int kk = k0 + koff + e + ((e >= 8) ? 8 : 0);
      bb[e] = (kk < N_T) ? qkv[((size_t)(b * N_T + kk)) * N_TD3 + vcol] : z;
    }
    c = wmma_bf16(a, bb, c);
  }
  const int m0 = (lane >> 4) * 8;
#pragma unroll
  for (int v = 0; v < 8; ++v) {
    int q = qt * 16 + m0 + v;
    if (q < N_T)
      ao[((size_t)(b * N_T + q)) * N_D + (size_t)h * N_DK + dt * 16 + r16] = c[v];
  }
}

__global__ void k_olens(const int* __restrict__ ilens, int* __restrict__ o) {
  int b = threadIdx.x;
  if (b < N_B) o[b] = ilens[b];
}

// ---------------------------------------------------------------------------
// Host orchestration
// ---------------------------------------------------------------------------
extern "C" void kernel_launch(void* const* d_in, const int* in_sizes, int n_in,
                              void* d_out, int out_size, void* d_ws, size_t ws_size,
                              hipStream_t stream) {
  (void)in_sizes; (void)n_in; (void)out_size; (void)ws_size;
  const float* xs    = (const float*)d_in[0];
  const int*   ilens = (const int*)d_in[1];
  const float* an_s  = (const float*)d_in[41];
  const float* an_b  = (const float*)d_in[42];
  const float* tn_s  = (const float*)d_in[43];
  const float* tn_b  = (const float*)d_in[44];

  char* cur = (char*)d_ws;
  auto alloc = [&](size_t bytes) -> void* {
    void* p = cur;
    cur += (bytes + 255) & ~((size_t)255);
    return p;
  };
  float*  xin    = (float*)alloc((size_t)N_M * N_DIN * 4);
  float*  x      = (float*)alloc((size_t)N_M * N_D * 4);
  __bf16* hbf    = (__bf16*)alloc((size_t)N_M * N_KPAD0 * 2);
  float*  qkv    = (float*)alloc((size_t)N_M * N_TD3 * 4);
  __bf16* qkvb   = (__bf16*)alloc((size_t)N_M * N_TD3 * 2);
  float*  scores = (float*)alloc((size_t)N_B * N_H * N_T * N_SLD * 4);
  __bf16* pbf    = (__bf16*)alloc((size_t)N_B * N_H * N_T * N_SLD * 2);
  float*  ao     = (float*)alloc((size_t)N_M * N_D * 4);
  __bf16* aobf   = (__bf16*)alloc((size_t)N_M * N_D * 2);
  float*  fs     = (float*)alloc((size_t)N_M * N_D * 4);
  float*  ffh    = (float*)alloc((size_t)N_M * N_F * 4);
  __bf16* ffhb   = (__bf16*)alloc((size_t)N_M * N_F * 2);
  __bf16* wqkv   = (__bf16*)alloc((size_t)N_TD3 * N_KPAD0 * 2);
  __bf16* wout   = (__bf16*)alloc((size_t)N_D * N_D * 2);
  __bf16* w1b    = (__bf16*)alloc((size_t)N_F * N_D * 2);
  __bf16* w2b    = (__bf16*)alloc((size_t)N_D * N_F * 2);

  k_posenc<<<(unsigned)(((size_t)N_M * N_DIN + 255) / 256), 256, 0, stream>>>(xs, xin);

  for (int li = 0; li < 10; ++li) {
    int base, off, din;
    if (li == 0)      { base = 2;  off = 0;      din = N_DIN; }
    else if (li <= 5) { base = 15; off = li - 1; din = N_D;   }
    else              { base = 28; off = li - 6; din = N_D;   }
    // param dict flattened with sorted keys:
    // b1,b2,fsmn_w,n1_b,n1_s,n2_b,n2_s,out_b,out_w,qkv_b,qkv_w,w1,w2
    const float* Ar[13];
    for (int i = 0; i < 13; ++i) Ar[i] = (const float*)d_in[base + i];
    size_t s = (size_t)off;
    const float* b1      = Ar[0]  + s * N_F;
    const float* b2      = Ar[1]  + s * N_D;
    const float* fsw     = Ar[2]  + s * ((size_t)N_D * N_KC);
    const float* n1b_    = Ar[3]  + s * din;
    const float* n1s_    = Ar[4]  + s * din;
    const float* n2b_    = Ar[5]  + s * N_D;
    const float* n2s_    = Ar[6]  + s * N_D;
    const float* outb    = Ar[7]  + s * N_D;
    const float* outw    = Ar[8]  + s * ((size_t)N_D * N_D);
    const float* qkvbias = Ar[9]  + s * N_TD3;
    const float* qkvw    = Ar[10] + s * ((size_t)N_TD3 * din);
    const float* w1      = Ar[11] + s * ((size_t)N_F * N_D);
    const float* w2      = Ar[12] + s * ((size_t)N_D * N_F);
    const int kpad = (li == 0) ? N_KPAD0 : N_D;
    const float* xl    = (li == 0) ? xin : x;
    const float* resid = (li == 0) ? nullptr : x;

    auto cw = [&](const float* wsrc, __bf16* wdst, int N, int Kin, int Kp) {
      size_t cnt = (size_t)N * Kp;
      k_convw<<<(unsigned)((cnt + 255) / 256), 256, 0, stream>>>(wsrc, wdst, N, Kin, Kp);
    };
    cw(qkvw, wqkv, N_TD3, din, kpad);
    cw(outw, wout, N_D, N_D, N_D);
    cw(w1,   w1b,  N_F, N_D, N_D);
    cw(w2,   w2b,  N_D, N_F, N_F);

    // h = LN1(x) -> bf16 (K padded)
    k_ln<<<N_M, 256, 0, stream>>>(xl, n1s_, n1b_, hbf, nullptr, din, kpad);
    // qkv = h * qkv_w^T + qkv_b
    k_gemm<<<dim3(N_TD3 / 32, N_M / 64), 128, 0, stream>>>(
        hbf, wqkv, qkvbias, nullptr, nullptr, qkv, N_TD3, kpad, 0);
    k_f2b<<<(unsigned)(((size_t)N_M * N_TD3 + 255) / 256), 256, 0, stream>>>(
        qkv, qkvb, (size_t)N_M * N_TD3);
    // fsmn branch from v
    k_fsmn<<<(unsigned)(((size_t)N_M * N_D + 255) / 256), 256, 0, stream>>>(
        qkv, fsw, ilens, fs);
    // attention
    k_scores<<<dim3(N_QT, N_QT, N_B * N_H), 32, 0, stream>>>(qkvb, scores);
    k_softmax<<<dim3(N_T, N_B * N_H), 256, 0, stream>>>(scores, ilens, pbf);
    k_av<<<dim3(N_QT, N_DK / 16, N_B * N_H), 32, 0, stream>>>(pbf, qkvb, ao);
    k_f2b<<<(unsigned)(((size_t)N_M * N_D + 255) / 256), 256, 0, stream>>>(
        ao, aobf, (size_t)N_M * N_D);
    // x = [x +] (o * out_w^T + out_b + fsmn)
    k_gemm<<<dim3(N_D / 32, N_M / 64), 128, 0, stream>>>(
        aobf, wout, outb, resid, fs, x, N_D, N_D, 0);
    // FFN
    k_ln<<<N_M, 256, 0, stream>>>(x, n2s_, n2b_, hbf, nullptr, N_D, N_D);
    k_gemm<<<dim3(N_F / 32, N_M / 64), 128, 0, stream>>>(
        hbf, w1b, b1, nullptr, nullptr, ffh, N_F, N_D, 1);
    k_f2b<<<(unsigned)(((size_t)N_M * N_F + 255) / 256), 256, 0, stream>>>(
        ffh, ffhb, (size_t)N_M * N_F);
    k_gemm<<<dim3(N_D / 32, N_M / 64), 128, 0, stream>>>(
        ffhb, w2b, b2, x, nullptr, x, N_D, N_F, 0);

    if (li == 5)  // after encoder stack: x = LN(x, an) in place
      k_ln<<<N_M, 256, 0, stream>>>(x, an_s, an_b, nullptr, x, N_D, N_D);
  }
  // final LN -> d_out, then olens (= ilens) as raw int32 bits
  k_ln<<<N_M, 256, 0, stream>>>(x, tn_s, tn_b, nullptr, (float*)d_out, N_D, N_D);
  k_olens<<<1, 32, 0, stream>>>(ilens, (int*)d_out + (size_t)N_M * N_D);
}
